// AtlasnetDecoder_66013647339788
// MI455X (gfx1250) — matrix operand order repaired and checked by reading, not compiled, not run on hardware
//
#include <hip/hip_runtime.h>
#include <hip/hip_bf16.h>
#include <math.h>

typedef __attribute__((ext_vector_type(16))) _Float16 v16h;
typedef __attribute__((ext_vector_type(8)))  _Float16 v8h;
typedef __attribute__((ext_vector_type(8)))  float    v8f;

// ---------------------------------------------------------------------------
// Fragment loaders per CDNA5 ISA 7.12.2 (wave32, 16-bit operands)
// A (16x32, MxK) row-major [row][k], ld in elements:
//   lane = hf*16 + r ; holds row M=r ; elems 0..7 = K in [hf*8, hf*8+8),
//   elems 8..15 = K in [16+hf*8, 16+hf*8+8)
// ---------------------------------------------------------------------------
__device__ __forceinline__ v16h load_fragA(const _Float16* base, int row0, int ld, int k0) {
    int lane = threadIdx.x & 31;
    int r = lane & 15, hf = lane >> 4;
    const _Float16* p = base + (size_t)(row0 + r) * ld + k0 + hf * 8;
    v8h lo = *(const v8h*)(p);
    v8h hi = *(const v8h*)(p + 16);
    v16h out;
#pragma unroll
    for (int i = 0; i < 8; ++i) { out[i] = lo[i]; out[8 + i] = hi[i]; }
    return out;
}

// B (32x16, KxN) with weights stored transposed as Wt[n][k], ld in elements:
//   lane = hf*16 + n ; holds column N=n ; elems 0..7 = K in [hf*8, hf*8+8),
//   elems 8..15 = K in [16+hf*8, 16+hf*8+8)
__device__ __forceinline__ v16h load_fragB(const _Float16* base, int n0, int ld, int k0) {
    int lane = threadIdx.x & 31;
    int n = lane & 15, hf = lane >> 4;
    const _Float16* p = base + (size_t)(n0 + n) * ld + k0 + hf * 8;
    v8h lo = *(const v8h*)(p);
    v8h hi = *(const v8h*)(p + 16);
    v16h out;
#pragma unroll
    for (int i = 0; i < 8; ++i) { out[i] = lo[i]; out[8 + i] = hi[i]; }
    return out;
}

// ---------------------------------------------------------------------------
// One fused GEMM layer: D(64xN) = relu(A(64xK) @ Wt(NxK)^T + bias) -> LDS f16
// 8 waves, each owning a disjoint N-range (NT 16-wide tiles) and covering all
// 4 M-strips: every B fragment is loaded from global exactly once per block.
// ---------------------------------------------------------------------------
template <int K, int NT>
__device__ __forceinline__ void gemm_layer(const _Float16* aLds, int aLd,
                                           const _Float16* w /*[N][K] patch base*/,
                                           const float* bias,
                                           _Float16* dLds, int dLd) {
    const int lane  = threadIdx.x & 31;
    const int wid   = threadIdx.x >> 5;   // 0..7
    const int nbase = wid * (NT * 16);    // wave-owned N range

    v8f c[4][NT] = {};
    for (int k0 = 0; k0 < K; k0 += 32) {
        v16h a[4];
#pragma unroll
        for (int s = 0; s < 4; ++s) a[s] = load_fragA(aLds, s * 16, aLd, k0);
#pragma unroll
        for (int i = 0; i < NT; ++i) {
            v16h b = load_fragB(w, nbase + i * 16, K, k0);
#pragma unroll
            for (int s = 0; s < 4; ++s)
                c[s][i] = __builtin_amdgcn_wmma_f32_16x16x32_f16(
                    false, a[s], false, b, (short)0, c[s][i], false, false);
        }
    }
    // C/D layout: vgpr r, lane -> M = r + 8*hf, N = lane&15
    const int hf = lane >> 4, nl = lane & 15;
#pragma unroll
    for (int i = 0; i < NT; ++i) {
        int n = nbase + i * 16 + nl;
        float bs = bias[n];
#pragma unroll
        for (int s = 0; s < 4; ++s) {
#pragma unroll
            for (int r = 0; r < 8; ++r) {
                int m = s * 16 + r + 8 * hf;
                dLds[m * dLd + n] = (_Float16)fmaxf(c[s][i][r] + bs, 0.0f);
            }
        }
    }
}

// ---------------------------------------------------------------------------
// Kernel 0: transpose+convert weights f32 [P][K][N] -> f16 [P][N][K]
// ---------------------------------------------------------------------------
__global__ void transpose_fp16(const float* __restrict__ in, _Float16* __restrict__ out,
                               int K, int N) {
    __shared__ float tile[32][33];
    int p = blockIdx.z;
    int n0 = blockIdx.x * 32, k0 = blockIdx.y * 32;
    int tx = threadIdx.x, ty = threadIdx.y; // 32 x 8
    const float* inp = in + (size_t)p * K * N;
#pragma unroll
    for (int i = 0; i < 4; ++i)
        tile[ty + i * 8][tx] = inp[(size_t)(k0 + ty + i * 8) * N + (n0 + tx)];
    __syncthreads();
    _Float16* outp = out + (size_t)p * N * K;
#pragma unroll
    for (int i = 0; i < 4; ++i)
        outp[(size_t)(n0 + ty + i * 8) * K + (k0 + tx)] = (_Float16)tile[tx][ty + i * 8];
}

// ---------------------------------------------------------------------------
// Kernel 1: T[p][b][n] = x[b] @ W1[p,:1024,:] + b1[p]   (memory-bound, W1 read once)
//           also extract W1u = W1[p,1024,:], W1v = W1[p,1025,:]
// grid = (25 patches, 4 column-slices of 256), block = 256 threads.
// ---------------------------------------------------------------------------
__global__ __launch_bounds__(256)
void compute_T(const float* __restrict__ x, const float* __restrict__ W1,
               const float* __restrict__ b1,
               float* __restrict__ T, float* __restrict__ w1u, float* __restrict__ w1v) {
    __shared__ float xs[16][64];
    const int p = blockIdx.x, t = threadIdx.x;
    const int col = blockIdx.y * 256 + t;

    float acc[16];
    const float bb = b1[p * 1024 + col];
#pragma unroll
    for (int b = 0; b < 16; ++b) acc[b] = bb;
    for (int k0 = 0; k0 < 1024; k0 += 64) {
        __syncthreads();
        for (int i = t; i < 16 * 64; i += 256) {
            int b = i >> 6, k = i & 63;
            xs[b][k] = x[b * 1024 + k0 + k];
        }
        __syncthreads();
        for (int k = 0; k < 64; ++k) {
            float wv = W1[((size_t)p * 1026 + (k0 + k)) * 1024 + col];
#pragma unroll
            for (int b = 0; b < 16; ++b) acc[b] = fmaf(xs[b][k], wv, acc[b]);
        }
    }
#pragma unroll
    for (int b = 0; b < 16; ++b)
        T[((size_t)p * 16 + b) * 1024 + col] = acc[b];

    w1u[p * 1024 + col] = W1[((size_t)p * 1026 + 1024) * 1024 + col];
    w1v[p * 1024 + col] = W1[((size_t)p * 1026 + 1025) * 1024 + col];
}

// ---------------------------------------------------------------------------
// Kernel 2: fused layers 1(point part) + 2 + 3 + 4 + 5 per (patch, 64-row tile)
// LDS overlay (padded rows, +8 halves to break 2KB bank aliasing):
//   region A: h1 64x(1024+8)  later h3 64x(256+8)
//   region B: h2 64x(512+8)   later h4 64x(128+8)
// ---------------------------------------------------------------------------
#define LD1 1032
#define LD2 520
#define LD3 264
#define LD4 136
#define REGA_BYTES (64 * LD1 * 2)               // 132096
#define SMEM_BYTES (REGA_BYTES + 64 * LD2 * 2)  // 198656

__global__ __launch_bounds__(256, 1)
void fused_mlp(const float* __restrict__ T, const float* __restrict__ w1u,
               const float* __restrict__ w1v,
               const _Float16* __restrict__ Wt2, const float* __restrict__ b2,
               const _Float16* __restrict__ Wt3, const float* __restrict__ b3,
               const _Float16* __restrict__ Wt4, const float* __restrict__ b4,
               const float* __restrict__ W5, const float* __restrict__ b5,
               float* __restrict__ out) {
    extern __shared__ char smem[];
    _Float16* bufA = (_Float16*)smem;                 // h1, then h3
    _Float16* bufB = (_Float16*)(smem + REGA_BYTES);  // h2, then h4

    const int p = blockIdx.x;          // patch 0..24
    const int row0 = blockIdx.y * 64;  // tile of 64 (b,g) rows, r = b*400 + g
    const int t = threadIdx.x;

    // ---- build h1 tile: relu(T[b,:] + u*W1u + v*W1v), 8 halves per iter ----
    const float* Tp = T + (size_t)p * 16 * 1024;
    const float* up = w1u + p * 1024;
    const float* vp = w1v + p * 1024;
    for (int q = t; q < 64 * 128; q += 256) {
        int m = q >> 7, kq = (q & 127) * 8;
        int r = row0 + m;
        int b = r / 400, g = r - b * 400;
        float u = (float)(g % 20) * (1.0f / 19.0f);
        float v = (float)(g / 20) * (1.0f / 19.0f);
        const float4* Tq = (const float4*)(Tp + b * 1024 + kq);
        const float4* uq = (const float4*)(up + kq);
        const float4* vq = (const float4*)(vp + kq);
        float4 T0 = Tq[0], T1 = Tq[1];
        float4 u0 = uq[0], u1 = uq[1];
        float4 v0 = vq[0], v1 = vq[1];
        v8h o;
        o[0] = (_Float16)fmaxf(fmaf(u, u0.x, fmaf(v, v0.x, T0.x)), 0.0f);
        o[1] = (_Float16)fmaxf(fmaf(u, u0.y, fmaf(v, v0.y, T0.y)), 0.0f);
        o[2] = (_Float16)fmaxf(fmaf(u, u0.z, fmaf(v, v0.z, T0.z)), 0.0f);
        o[3] = (_Float16)fmaxf(fmaf(u, u0.w, fmaf(v, v0.w, T0.w)), 0.0f);
        o[4] = (_Float16)fmaxf(fmaf(u, u1.x, fmaf(v, v1.x, T1.x)), 0.0f);
        o[5] = (_Float16)fmaxf(fmaf(u, u1.y, fmaf(v, v1.y, T1.y)), 0.0f);
        o[6] = (_Float16)fmaxf(fmaf(u, u1.z, fmaf(v, v1.z, T1.z)), 0.0f);
        o[7] = (_Float16)fmaxf(fmaf(u, u1.w, fmaf(v, v1.w, T1.w)), 0.0f);
        *(v8h*)(bufA + m * LD1 + kq) = o;
    }
    __syncthreads();

    // ---- layer 2: (64x1024) x (512x1024)^T -> h2 (64x512) ----
    gemm_layer<1024, 4>(bufA, LD1, Wt2 + (size_t)p * 512 * 1024, b2 + p * 512, bufB, LD2);
    __syncthreads();
    // ---- layer 3: (64x512) x (256x512)^T -> h3 (64x256) ----
    gemm_layer<512, 2>(bufB, LD2, Wt3 + (size_t)p * 256 * 512, b3 + p * 256, bufA, LD3);
    __syncthreads();
    // ---- layer 4: (64x256) x (128x256)^T -> h4 (64x128) ----
    gemm_layer<256, 1>(bufA, LD3, Wt4 + (size_t)p * 128 * 256, b4 + p * 128, bufB, LD4);
    __syncthreads();

    // ---- layer 5 (N=3) + tanh, VALU ----
    if (t < 192) {
        int m = t / 3, c = t - (t / 3) * 3;
        const float* W5p = W5 + (size_t)p * 128 * 3;
        float acc = b5[p * 3 + c];
        for (int k = 0; k < 128; ++k)
            acc = fmaf((float)bufB[m * LD4 + k], W5p[k * 3 + c], acc);
        int r = row0 + m;
        int b = r / 400, g = r - b * 400;
        out[(((size_t)b * 25 + p) * 400 + g) * 3 + c] = tanhf(acc);
    }
}

// ---------------------------------------------------------------------------
extern "C" void kernel_launch(void* const* d_in, const int* in_sizes, int n_in,
                              void* d_out, int out_size, void* d_ws, size_t ws_size,
                              hipStream_t stream) {
    const float* x  = (const float*)d_in[0];
    const float* W1 = (const float*)d_in[1];
    const float* b1 = (const float*)d_in[2];
    const float* W2 = (const float*)d_in[3];
    const float* b2 = (const float*)d_in[4];
    const float* W3 = (const float*)d_in[5];
    const float* b3 = (const float*)d_in[6];
    const float* W4 = (const float*)d_in[7];
    const float* b4 = (const float*)d_in[8];
    const float* W5 = (const float*)d_in[9];
    const float* b5 = (const float*)d_in[10];
    float* out = (float*)d_out;

    char* ws = (char*)d_ws;
    _Float16* Wt2 = (_Float16*)(ws);                 // 25*512*1024*2  = 26,214,400
    _Float16* Wt3 = (_Float16*)(ws + 26214400);      // 25*256*512*2  =  6,553,600
    _Float16* Wt4 = (_Float16*)(ws + 32768000);      // 25*128*256*2  =  1,638,400
    float*    T   = (float*)   (ws + 34406400);      // 25*16*1024*4  =  1,638,400
    float*    w1u = (float*)   (ws + 36044800);      // 25*1024*4     =    102,400
    float*    w1v = (float*)   (ws + 36147200);      // total 36,249,600 bytes

    dim3 tb(32, 8);
    transpose_fp16<<<dim3(512 / 32, 1024 / 32, 25), tb, 0, stream>>>(W2, Wt2, 1024, 512);
    transpose_fp16<<<dim3(256 / 32, 512 / 32, 25), tb, 0, stream>>>(W3, Wt3, 512, 256);
    transpose_fp16<<<dim3(128 / 32, 256 / 32, 25), tb, 0, stream>>>(W4, Wt4, 256, 128);
    compute_T<<<dim3(25, 4), 256, 0, stream>>>(x, W1, b1, T, w1u, w1v);
    fused_mlp<<<dim3(25, 100), 256, SMEM_BYTES, stream>>>(
        T, w1u, w1v, Wt2, b2, Wt3, b3, Wt4, b4, W5, b5, out);
}
